// LinearCausalAttention_19232863552148
// MI455X (gfx1250) — compile-verified, matrix operand order checked by reading, not run on hardware
//
#include <hip/hip_runtime.h>
#include <hip/hip_bf16.h>

typedef __attribute__((ext_vector_type(16))) _Float16 v16h;
typedef __attribute__((ext_vector_type(8)))  _Float16 v8h;
typedef __attribute__((ext_vector_type(8)))  float    v8f;
typedef __attribute__((ext_vector_type(4)))  _Float16 v4h;
typedef __attribute__((ext_vector_type(4)))  int      v4i;

#define AS1 __attribute__((address_space(1)))
#define AS3 __attribute__((address_space(3)))

#define BB 2
#define TT 1024
#define CC 1024
#define HH 16
#define DD 64

// ---------------------------------------------------------------------------
// Async copy to LDS (CDNA5 ASYNCcnt path), guarded so compile cannot fail.
// Probe-verified signature: (v4i AS1*, v4i AS3*, imm offset, imm cpol).
// ---------------------------------------------------------------------------
#if __has_builtin(__builtin_amdgcn_global_load_async_to_lds_b128)
#define HAVE_ASYNC 1
#else
#define HAVE_ASYNC 0
#endif

// copy one 16-byte chunk (8 halves) global -> LDS
static __device__ __forceinline__ void stage_chunk(_Float16* dst,
                                                   const _Float16* src) {
#if HAVE_ASYNC
  __builtin_amdgcn_global_load_async_to_lds_b128(
      (AS1 v4i*)src, (AS3 v4i*)dst, 0, 0);
#else
  *(uint4*)dst = *(const uint4*)src;
#endif
}

#if HAVE_ASYNC
#if __has_builtin(__builtin_amdgcn_s_wait_asynccnt)
#define WAIT_STAGE(n) __builtin_amdgcn_s_wait_asynccnt(n)
#else
#define WAIT_STAGE(n) asm volatile("s_wait_asynccnt %0" ::"i"(n) : "memory")
#endif
#else
#define WAIT_STAGE(n) do {} while (0)
#endif

// ---------------------------------------------------------------------------
// WMMA helpers (gfx1250 wave32).  D = A(16x32 f16) x B(32x16 f16) + C(f32).
// A fragment: lane L in [0,16) holds row M=L; K = (L>=16 ? 8:0) + {0..7} in
// VGPR0..3 and +16 in VGPR4..7 (ISA 7.12.2).  B fragment follows the same
// pattern with N instead of M, i.e. it reads contiguous rows of B-transposed.
// ---------------------------------------------------------------------------
static __device__ __forceinline__ v8f wmma16(v16h a, v16h b, v8f c) {
  return __builtin_amdgcn_wmma_f32_16x16x32_f16(false, a, false, b, (short)0, c,
                                                false, false);
}

static __device__ __forceinline__ v16h frag_row(const _Float16* __restrict__ p0,
                                                int ld, int lane) {
  const int m    = lane & 15;
  const int koff = (lane >> 4) << 3;
  const _Float16* p = p0 + m * ld + koff;
  v16h r;
#pragma unroll
  for (int j = 0; j < 4; ++j) {
    r[2 * j]     = p[2 * j];
    r[2 * j + 1] = p[2 * j + 1];
    r[2 * j + 8] = p[16 + 2 * j];
    r[2 * j + 9] = p[17 + 2 * j];
  }
  return r;
}

// ---------------------------------------------------------------------------
// f32 -> f16 pack (one pass; everything is L2-resident at 23.3 TB/s)
// ---------------------------------------------------------------------------
__global__ __launch_bounds__(256) void cvt_f32_f16(const float* __restrict__ in,
                                                   _Float16* __restrict__ out,
                                                   int n4) {
  const int i = blockIdx.x * 256 + threadIdx.x;
  if (i < n4) {
    float4 f = ((const float4*)in)[i];
    v4h h;
    h[0] = (_Float16)f.x; h[1] = (_Float16)f.y;
    h[2] = (_Float16)f.z; h[3] = (_Float16)f.w;
    ((v4h*)out)[i] = h;
  }
}

// ---------------------------------------------------------------------------
// GEMM: out[M,N] = A[M,K] * W[N,K]^T + bias, A/W pre-packed f16.
// 128x64 block tile, BK=64, double-buffered async LDS staging.  8 waves in a
// 4x2 grid; each wave owns a 2x2 set of 16x16 WMMA tiles (8 wmma / 8 frag
// loads per K-step).  MODE 0: plain f32 store.  MODE 1: fused qkv epilogue
// (elu+1 on q/k thirds, scatter q',k',v as f16 into (B,H,T,D) head layout).
// ---------------------------------------------------------------------------
template <int MODE>
__global__ __launch_bounds__(256) void gemm_f16wmma(
    const _Float16* __restrict__ A, const _Float16* __restrict__ W,
    const float* __restrict__ bias, float* __restrict__ outF,
    _Float16* __restrict__ qp, _Float16* __restrict__ kp,
    _Float16* __restrict__ vv, int M, int N, int K) {
  __shared__ _Float16 As[2][128 * 64];
  __shared__ _Float16 Ws[2][64 * 64];
  const int tid  = threadIdx.x;
  const int lane = tid & 31, wid = tid >> 5;
  const int wr = wid & 3, wc = wid >> 2;   // 4x2 wave grid
  const int m0 = blockIdx.y * 128, n0 = blockIdx.x * 64;

  // staging: A tile = 1024 16B-chunks (4/thread), W tile = 512 (2/thread)
  auto stage = [&](int buf, int k0) {
#pragma unroll
    for (int j = 0; j < 4; ++j) {
      const int l = tid * 4 + j;
      const int row = l >> 3, sg = (l & 7) * 8;
      stage_chunk(&As[buf][row * 64 + sg], A + (m0 + row) * K + k0 + sg);
    }
#pragma unroll
    for (int j = 0; j < 2; ++j) {
      const int l = tid * 2 + j;
      const int row = l >> 3, sg = (l & 7) * 8;
      stage_chunk(&Ws[buf][row * 64 + sg], W + (n0 + row) * K + k0 + sg);
    }
  };

  v8f acc[2][2] = {{{}, {}}, {{}, {}}};
  const int steps = K >> 6;
  stage(0, 0);
  for (int it = 0; it < steps; ++it) {
    if (it + 1 < steps) {
      stage((it + 1) & 1, (it + 1) << 6);  // prefetch next buffer
      WAIT_STAGE(6);                       // keep next stage in flight
    } else {
      WAIT_STAGE(0);
    }
    __syncthreads();
    const _Float16* as = As[it & 1];
    const _Float16* ws = Ws[it & 1];
#pragma unroll
    for (int kk = 0; kk < 64; kk += 32) {
      v16h a0 = frag_row(as + (wr * 32) * 64 + kk, 64, lane);
      v16h a1 = frag_row(as + (wr * 32 + 16) * 64 + kk, 64, lane);
      v16h b0 = frag_row(ws + (wc * 32) * 64 + kk, 64, lane);
      v16h b1 = frag_row(ws + (wc * 32 + 16) * 64 + kk, 64, lane);
      acc[0][0] = wmma16(a0, b0, acc[0][0]);
      acc[0][1] = wmma16(a0, b1, acc[0][1]);
      acc[1][0] = wmma16(a1, b0, acc[1][0]);
      acc[1][1] = wmma16(a1, b1, acc[1][1]);
    }
    __syncthreads();
  }

  const int hi = lane >> 4, ln = lane & 15;
#pragma unroll
  for (int sr = 0; sr < 2; ++sr) {
#pragma unroll
    for (int sc = 0; sc < 2; ++sc) {
      const int ncol = n0 + wc * 32 + sc * 16 + ln;
      const float bv = bias[ncol];
#pragma unroll
      for (int r = 0; r < 8; ++r) {
        const int mrow = m0 + wr * 32 + sr * 16 + hi * 8 + r;
        float val      = acc[sr][sc][r] + bv;
        if (MODE == 0) {
          outF[mrow * N + ncol] = val;
        } else {
          const int part = ncol >> 10;        // 0=q 1=k 2=v   (C == 1024)
          const int cc   = ncol & 1023;
          const int h = cc >> 6, d = cc & 63; // D == 64
          const int b = mrow >> 10, t = mrow & 1023;
          const int off = (((b * HH + h) * TT) + t) * DD + d;
          if (part == 2) {
            vv[off] = (_Float16)val;
          } else {
            val = val > 0.f ? val + 1.f : __expf(val);  // elu(x)+1
            if (part == 0) qp[off] = (_Float16)val;
            else           kp[off] = (_Float16)val;
          }
        }
      }
    }
  }
}

// ---------------------------------------------------------------------------
// Chunked causal linear attention.  One workgroup per (b,h); 16 sequential
// 64-step chunks.  State S^T (64x64) lives in registers across chunks and is
// mirrored to LDS f16 once per chunk.  Output written f16 for GEMM-3.
// ---------------------------------------------------------------------------
__global__ __launch_bounds__(256) void linattn_chunk(
    const _Float16* __restrict__ qp, const _Float16* __restrict__ kp,
    const _Float16* __restrict__ vv, _Float16* __restrict__ attnh) {
  __shared__ _Float16 Qc[64 * 64];    // q'  (t, d)
  __shared__ _Float16 Kc[64 * 64];    // k'  (t, d)
  __shared__ _Float16 Kt[64 * 64];    // k'^T (d, t)
  __shared__ _Float16 Vt[64 * 64];    // v^T  (e, t)
  __shared__ _Float16 Am[64 * 64];    // masked scores (t, t)
  __shared__ _Float16 St16[64 * 64];  // state S^T f16 mirror (e, d)
  __shared__ float ksum[64];
  __shared__ float nrm[64];           // reciprocal normalizer

  const int tid  = threadIdx.x;
  const int lane = tid & 31, wid = tid >> 5;
  const int wr = wid & 3, wc = wid >> 2;
  const int hi = lane >> 4, ln = lane & 15;
  const int bh = blockIdx.x;          // b*H + h
  const int b = bh >> 4, h = bh & 15;

  const _Float16* qg = qp + (size_t)bh * TT * DD;
  const _Float16* kg = kp + (size_t)bh * TT * DD;
  const _Float16* vg = vv + (size_t)bh * TT * DD;

  v8f sacc[2] = {{}, {}};             // running S^T tiles
  if (tid < 64) ksum[tid] = 0.f;

  for (int c0 = 0; c0 < TT; c0 += 64) {
    // (a) async-stage q'/k'; mirror state; build k'^T and v^T
#pragma unroll
    for (int j = 0; j < 2; ++j) {
      const int off = (tid * 2 + j) * 8;  // 512 chunks of 8 halves
      stage_chunk(&Qc[off], qg + c0 * DD + off);
      stage_chunk(&Kc[off], kg + c0 * DD + off);
    }
#pragma unroll
    for (int s = 0; s < 2; ++s) {
      const int col = wc * 32 + s * 16 + ln;
#pragma unroll
      for (int r = 0; r < 8; ++r)
        St16[(wr * 16 + hi * 8 + r) * 64 + col] = (_Float16)sacc[s][r];
    }
    for (int i = tid; i < 512; i += 256) {
      const int t = i >> 3, d0 = (i & 7) * 8;
      union { uint4 u; _Float16 h[8]; } uv, uk;
      uv.u = *(const uint4*)(vg + c0 * DD + t * 64 + d0);
      uk.u = *(const uint4*)(kg + c0 * DD + t * 64 + d0);
#pragma unroll
      for (int d = 0; d < 8; ++d) {
        Vt[(d0 + d) * 64 + t] = uv.h[d];
        Kt[(d0 + d) * 64 + t] = uk.h[d];
      }
    }
    WAIT_STAGE(0);
    __syncthreads();

    // (b) scores A = Q' K'^T with causal mask, stored f16
#pragma unroll
    for (int s = 0; s < 2; ++s) {
      const int cj = wc * 2 + s;
      if (cj > wr) {                        // fully above the diagonal
#pragma unroll
        for (int r = 0; r < 8; ++r)
          Am[(wr * 16 + hi * 8 + r) * 64 + cj * 16 + ln] = (_Float16)0.f;
      } else {
        v8f sc = {};
#pragma unroll
        for (int k0 = 0; k0 < 64; k0 += 32) {
          v16h aq = frag_row(Qc + (wr * 16) * 64 + k0, 64, lane);
          v16h bk = frag_row(Kc + (cj * 16) * 64 + k0, 64, lane);
          sc = wmma16(aq, bk, sc);
        }
#pragma unroll
        for (int r = 0; r < 8; ++r) {
          const int m = hi * 8 + r;
          float v = sc[r];
          if (cj == wr && ln > m) v = 0.f;  // diagonal tile mask
          Am[(wr * 16 + m) * 64 + cj * 16 + ln] = (_Float16)v;
        }
      }
    }
    __syncthreads();

    // (c) reciprocal normalizer: 1/(rowsum(masked A) + q'.ksum_prev + eps)
    if (tid < 64) {
      float sN = 0.f;
      const v8h* am8 = (const v8h*)(Am + tid * 64);
      const v8h* qc8 = (const v8h*)(Qc + tid * 64);
#pragma unroll
      for (int j = 0; j < 8; ++j) {
        v8h a = am8[j];
        v8h q = qc8[j];
#pragma unroll
        for (int e = 0; e < 8; ++e) {
          sN += (float)a[e];
          sN += (float)q[e] * ksum[j * 8 + e];
        }
      }
      nrm[tid] = 1.0f / (sN + 1e-6f);
    }
    __syncthreads();

    // (d) advance running k-sum via contiguous k'^T rows
    if (tid < 64) {
      float sK = ksum[tid];
      const v8h* kt8 = (const v8h*)(Kt + tid * 64);
#pragma unroll
      for (int j = 0; j < 8; ++j) {
        v8h kv = kt8[j];
#pragma unroll
        for (int e = 0; e < 8; ++e) sK += (float)kv[e];
      }
      ksum[tid] = sK;
    }

    // (e) context = A@V + Q'@S ; multiply by reciprocal ; store f16
#pragma unroll
    for (int s = 0; s < 2; ++s) {
      const int cn = wc * 32 + s * 16;       // output e-tile origin
      v8f ctx = {};
#pragma unroll
      for (int k0 = 0; k0 < 64; k0 += 32) {
        v16h aa = frag_row(Am + (wr * 16) * 64 + k0, 64, lane);
        v16h bv = frag_row(Vt + cn * 64 + k0, 64, lane);     // V^T rows = B cols
        ctx = wmma16(aa, bv, ctx);
      }
#pragma unroll
      for (int k0 = 0; k0 < 64; k0 += 32) {
        v16h aq = frag_row(Qc + (wr * 16) * 64 + k0, 64, lane);
        v16h bs = frag_row(St16 + cn * 64 + k0, 64, lane);   // S^T rows = B cols
        ctx = wmma16(aq, bs, ctx);
      }
#pragma unroll
      for (int r = 0; r < 8; ++r) {
        const int tl = wr * 16 + hi * 8 + r;
        attnh[((size_t)(b * TT + c0 + tl)) * CC + h * DD + cn + ln] =
            (_Float16)(ctx[r] * nrm[tl]);
      }
    }

    // (f) state update: S^T += V^T @ K'  (B operand from contiguous k'^T)
#pragma unroll
    for (int s = 0; s < 2; ++s) {
      const int cd = wc * 32 + s * 16;
#pragma unroll
      for (int k0 = 0; k0 < 64; k0 += 32) {
        v16h av = frag_row(Vt + (wr * 16) * 64 + k0, 64, lane);
        v16h bk = frag_row(Kt + cd * 64 + k0, 64, lane);
        sacc[s] = wmma16(av, bk, sacc[s]);
      }
    }
    __syncthreads();   // protect LDS before next chunk overwrites
  }
}

// ---------------------------------------------------------------------------
extern "C" void kernel_launch(void* const* d_in, const int* in_sizes, int n_in,
                              void* d_out, int out_size, void* d_ws,
                              size_t ws_size, hipStream_t stream) {
  (void)in_sizes; (void)n_in; (void)out_size; (void)ws_size;
  const float* x     = (const float*)d_in[0];
  const float* w_qkv = (const float*)d_in[1];
  const float* b_qkv = (const float*)d_in[2];
  const float* w_out = (const float*)d_in[3];
  const float* b_out = (const float*)d_in[4];
  float* out = (float*)d_out;

  const int M = BB * TT;                            // 2048 rows
  const size_t nX  = (size_t)M * CC;                // 2M
  const size_t nWq = (size_t)3 * CC * CC;           // 3M
  const size_t nWo = (size_t)CC * CC;               // 1M
  const size_t nHD = (size_t)BB * HH * TT * DD;     // 2M

  _Float16* xh    = (_Float16*)d_ws;
  _Float16* wqh   = xh + nX;
  _Float16* woh   = wqh + nWq;
  _Float16* qp    = woh + nWo;
  _Float16* kp    = qp + nHD;
  _Float16* vv    = kp + nHD;
  _Float16* attnh = vv + nHD;                       // total ~28 MB

  dim3 blk(256);
  cvt_f32_f16<<<dim3((int)(nX  / 1024)), blk, 0, stream>>>(x, xh, (int)(nX / 4));
  cvt_f32_f16<<<dim3((int)(nWq / 1024)), blk, 0, stream>>>(w_qkv, wqh, (int)(nWq / 4));
  cvt_f32_f16<<<dim3((int)(nWo / 1024)), blk, 0, stream>>>(w_out, woh, (int)(nWo / 4));

  // QKV projection + fused elu()+1 + head repack (f16 out)
  gemm_f16wmma<1><<<dim3(3 * CC / 64, M / 128), blk, 0, stream>>>(
      xh, wqh, b_qkv, nullptr, qp, kp, vv, M, 3 * CC, CC);
  // chunked causal linear attention, one WG per (b,h)
  linattn_chunk<<<dim3(BB * HH), blk, 0, stream>>>(qp, kp, vv, attnh);
  // output projection, f32 result
  gemm_f16wmma<0><<<dim3(CC / 64, M / 128), blk, 0, stream>>>(
      attnh, woh, b_out, out, nullptr, nullptr, nullptr, M, CC, CC);
}
